// Attention_14568529068294
// MI455X (gfx1250) — matrix operand order repaired and checked by reading, not compiled, not run on hardware
//
#include <hip/hip_runtime.h>
#include <hip/hip_bf16.h>
#include <stdint.h>

// Problem sizes (compile-time constants from the reference)
#define B_  2
#define L_  2048
#define D_  1024
#define H_  16
#define DH_ 64
#define M_  (B_ * L_)

typedef __attribute__((ext_vector_type(16))) __bf16   v16bf;
typedef __attribute__((ext_vector_type(8)))  float    v8f;
typedef __attribute__((ext_vector_type(8)))  uint32_t v8u;

// ---------------------------------------------------------------------------
// WMMA helpers (gfx1250 wave32, v_wmma_f32_16x16x32_bf16)
// ---------------------------------------------------------------------------
static __device__ __forceinline__ v8f wmma_bf16(v16bf a, v16bf b, v8f c) {
  return __builtin_amdgcn_wmma_f32_16x16x32_bf16(false, a, false, b, (short)0, c,
                                                 false, false);
}

static __device__ __forceinline__ v8f zero8() {
  v8f z;
#pragma unroll
  for (int r = 0; r < 8; ++r) z[r] = 0.0f;
  return z;
}

// A-matrix (16x32 bf16) per-lane k-base for VGPR j, given lane half:
//  lanes 0-15 : V0..V3 = K{0,1..6,7},  V4..V7 = K{16,17..22,23}
//  lanes16-31 : V0..V3 = K{8,9..14,15},V4..V7 = K{24,25..30,31}
static __device__ __forceinline__ int a_kbase(int j, int half) {
  return (j < 4 ? 2 * j : 16 + 2 * (j - 4)) + 8 * half;
}

static __device__ __forceinline__ v16bf load_afrag(const uint32_t* rowp, int half) {
  v8u u;
#pragma unroll
  for (int j = 0; j < 8; ++j) u[j] = rowp[a_kbase(j, half) >> 1];
  return __builtin_bit_cast(v16bf, u);
}

// B-matrix (32x16 bf16): same k order as A, Vj = K{a_kbase(j,half), +1}.
static __device__ __forceinline__ v16bf load_bfrag(const uint32_t* rowp, int half) {
  v8u u;
#pragma unroll
  for (int j = 0; j < 8; ++j) u[j] = rowp[a_kbase(j, half) >> 1];
  return __builtin_bit_cast(v16bf, u);
}

static __device__ __forceinline__ uint32_t pack_bf16(float lo, float hi) {
  uint16_t a = __builtin_bit_cast(uint16_t, (__bf16)lo);
  uint16_t b = __builtin_bit_cast(uint16_t, (__bf16)hi);
  return (uint32_t)a | ((uint32_t)b << 16);
}

// CDNA5 async global->LDS 16B copy (ASYNCcnt-tracked; no VGPR roundtrip).
static __device__ __forceinline__ void async_copy_b128(void* lds_dst,
                                                       const void* gsrc) {
  uint32_t ldsa = (uint32_t)(size_t)lds_dst;          // low 32 bits = LDS offset
  unsigned long long ga = (unsigned long long)(size_t)gsrc;
  asm volatile("global_load_async_to_lds_b128 %0, %1, off"
               :: "v"(ldsa), "v"(ga) : "memory");
}
static __device__ __forceinline__ void async_wait0() {
  asm volatile("s_wait_asynccnt 0x0" ::: "memory");
}

// ---------------------------------------------------------------------------
// Kernel 1: QKV projection.  C[M=4096, N=1024] = X[4096,1024] * W[1024,1024]
// (z selects Wq/Wk/Wv).  Output bf16, laid out [B,H,L,Dh]; q scaled by 1/8.
// ---------------------------------------------------------------------------
__global__ __launch_bounds__(256) void qkv_kernel(
    const float* __restrict__ x, const float* __restrict__ Wq,
    const float* __restrict__ Wk, const float* __restrict__ Wv,
    __bf16* __restrict__ qb, __bf16* __restrict__ kb, __bf16* __restrict__ vb) {
  const int z = blockIdx.z;
  const float* W   = (z == 0) ? Wq : (z == 1) ? Wk : Wv;
  __bf16*      out = (z == 0) ? qb : (z == 1) ? kb : vb;
  const float  scl = (z == 0) ? 0.125f : 1.0f;  // fold 1/sqrt(Dh) into q

  const int N0 = blockIdx.x * 64;
  const int M0 = blockIdx.y * 128;
  const int tid  = threadIdx.x;
  const int lane = tid & 31, wave = tid >> 5;
  const int half = lane >> 4, l16 = lane & 15;
  const int m0w = (wave >> 1) * 32;
  const int n0w = (wave & 1) * 32;

  __shared__ uint32_t As[128][16];    // 128 x 32 bf16 as packed k-pairs
  __shared__ __bf16   WsT[64][32];    // k-transposed W tile: [n][k]

  v8f acc[2][2];
  acc[0][0] = zero8(); acc[0][1] = zero8();
  acc[1][0] = zero8(); acc[1][1] = zero8();

  for (int kt = 0; kt < D_; kt += 32) {
    // Stage X tile (f32 -> bf16 pairs; conversion needs VALU, so no async)
#pragma unroll
    for (int i = 0; i < 4; ++i) {
      int idx4 = tid + i * 256;
      int row  = idx4 >> 3;
      int col  = (idx4 & 7) * 4;
      float4 f = *(const float4*)(x + (size_t)(M0 + row) * D_ + kt + col);
      As[row][(col >> 1) + 0] = pack_bf16(f.x, f.y);
      As[row][(col >> 1) + 1] = pack_bf16(f.z, f.w);
    }
    // Stage W tile transposed (f32 -> bf16)
#pragma unroll
    for (int i = 0; i < 2; ++i) {
      int idx4 = tid + i * 256;
      int kk   = idx4 >> 4;
      int n    = (idx4 & 15) * 4;
      float4 f = *(const float4*)(W + (size_t)(kt + kk) * D_ + N0 + n);
      WsT[n + 0][kk] = (__bf16)f.x;
      WsT[n + 1][kk] = (__bf16)f.y;
      WsT[n + 2][kk] = (__bf16)f.z;
      WsT[n + 3][kk] = (__bf16)f.w;
    }
    __syncthreads();

    v16bf a0 = load_afrag(&As[m0w + l16][0], half);
    v16bf a1 = load_afrag(&As[m0w + 16 + l16][0], half);
    v16bf b0 = load_bfrag((const uint32_t*)&WsT[n0w + l16][0], half);
    v16bf b1 = load_bfrag((const uint32_t*)&WsT[n0w + 16 + l16][0], half);
    acc[0][0] = wmma_bf16(a0, b0, acc[0][0]);
    acc[0][1] = wmma_bf16(a0, b1, acc[0][1]);
    acc[1][0] = wmma_bf16(a1, b0, acc[1][0]);
    acc[1][1] = wmma_bf16(a1, b1, acc[1][1]);
    __syncthreads();
  }

#pragma unroll
  for (int i = 0; i < 2; ++i)
#pragma unroll
    for (int jt = 0; jt < 2; ++jt)
#pragma unroll
      for (int r = 0; r < 8; ++r) {
        int gm = M0 + m0w + i * 16 + r + 8 * half;
        int gn = N0 + n0w + jt * 16 + l16;
        int b = gm >> 11, l = gm & (L_ - 1);
        int h = gn >> 6,  e = gn & (DH_ - 1);
        out[(((size_t)(b * H_ + h)) * L_ + l) * DH_ + e] =
            (__bf16)(acc[i][jt][r] * scl);
      }
}

// ---------------------------------------------------------------------------
// Kernel 2: causal flash attention per (b,h), double-buffered KV streaming.
// K chunks staged with global_load_async_to_lds_b128 (ASYNCcnt); V chunks
// transposed through VGPRs.  One barrier per chunk.
// ---------------------------------------------------------------------------
__global__ __launch_bounds__(256) void attn_kernel(
    const __bf16* __restrict__ qb, const __bf16* __restrict__ kb,
    const __bf16* __restrict__ vb, __bf16* __restrict__ ob) {
  const int qblk = blockIdx.x;
  const int bh   = blockIdx.y;
  const int tid  = threadIdx.x;
  const int lane = tid & 31, wave = tid >> 5;
  const int half = lane >> 4, l16 = lane & 15;
  const int q0   = qblk * 128 + wave * 16;
  const size_t base = (size_t)bh * L_ * DH_;

  __shared__ uint32_t Ks[2][32][32];    // [buf][kv][feat-pair], memory order
  __shared__ __bf16   VsT[2][64][32];   // [buf][feat][kv] (kv-transposed)
  __shared__ __bf16   Pw[8][16][32];    // per-wave P staging (C->A relayout)

  const int row = tid >> 3;
  const int fe  = (tid & 7) * 8;

  // Q fragments (feats 0..31 and 32..63), resident for whole KV sweep
  const uint32_t* qrow =
      (const uint32_t*)(qb + base + (size_t)(q0 + l16) * DH_);
  v16bf qf0, qf1;
  {
    v8u u0, u1;
#pragma unroll
    for (int j = 0; j < 8; ++j) {
      int k0 = a_kbase(j, half);
      u0[j] = qrow[k0 >> 1];
      u1[j] = qrow[(k0 + 32) >> 1];
    }
    qf0 = __builtin_bit_cast(v16bf, u0);
    qf1 = __builtin_bit_cast(v16bf, u1);
  }

  v8f o[4];
#pragma unroll
  for (int d = 0; d < 4; ++d) o[d] = zero8();
  float mrow[8], lrow[8];
#pragma unroll
  for (int r = 0; r < 8; ++r) { mrow[r] = -1e30f; lrow[r] = 0.0f; }

  // Stage one 32-row KV chunk into LDS buffer `buf`.
  auto stage = [&](int buf, int kv0) {
    // K: raw 16B copy, async straight into LDS
    async_copy_b128(&Ks[buf][row][fe >> 1],
                    kb + base + (size_t)(kv0 + row) * DH_ + fe);
    // V: needs kv-transpose -> VGPR roundtrip
    uint4 vd = *(const uint4*)(vb + base + (size_t)(kv0 + row) * DH_ + fe);
    union { uint4 u; __bf16 h[8]; } uv; uv.u = vd;
#pragma unroll
    for (int c = 0; c < 8; ++c) VsT[buf][fe + c][row] = uv.h[c];
  };

  const int kvend = qblk * 128 + 128;   // block-uniform trip count
  stage(0, 0);

  for (int c = 0, kv0 = 0; kv0 < kvend; ++c, kv0 += 32) {
    const int buf = c & 1;
    // Publish buffer `buf`: drain our async K writes, then block barrier.
    // The same barrier proves every wave finished reading buffer buf^1,
    // so prefetching into it below is race-free.
    async_wait0();
    __syncthreads();
    if (kv0 + 32 < kvend) stage(buf ^ 1, kv0 + 32);

    const bool active = (kv0 <= q0 + 15);   // wave-uniform: EXEC stays all-1s
    if (active) {
      // S = Q * K^T  (two 16x16 tiles, each = 2 chained WMMAs over Dh)
      v8f s[2];
#pragma unroll
      for (int t = 0; t < 2; ++t) {
        const uint32_t* krow = &Ks[buf][t * 16 + l16][0];
        v16bf kf0 = load_bfrag(krow, half);        // feats 0..31
        v16bf kf1 = load_bfrag(krow + 16, half);   // feats 32..63
        v8f cacc = zero8();
        cacc = wmma_bf16(qf0, kf0, cacc);
        cacc = wmma_bf16(qf1, kf1, cacc);
        s[t] = cacc;
      }

      // Causal mask + online softmax (rows map to vgpr r + 8*half)
      float mnew[8], alpha[8];
#pragma unroll
      for (int r = 0; r < 8; ++r) {
        int rowg = q0 + r + 8 * half;
#pragma unroll
        for (int t = 0; t < 2; ++t) {
          int colg = kv0 + t * 16 + l16;
          if (colg > rowg) s[t][r] = -1e30f;
        }
        float mx = fmaxf(s[0][r], s[1][r]);
        mx = fmaxf(mx, __shfl_xor(mx, 1, 32));
        mx = fmaxf(mx, __shfl_xor(mx, 2, 32));
        mx = fmaxf(mx, __shfl_xor(mx, 4, 32));
        mx = fmaxf(mx, __shfl_xor(mx, 8, 32));
        float mn = fmaxf(mrow[r], mx);
        alpha[r] = __expf(mrow[r] - mn);
        mnew[r]  = mn;
      }

#pragma unroll
      for (int r = 0; r < 8; ++r) {
        float p0 = __expf(s[0][r] - mnew[r]);
        float p1 = __expf(s[1][r] - mnew[r]);
        float rs = p0 + p1;
        rs += __shfl_xor(rs, 1, 32);
        rs += __shfl_xor(rs, 2, 32);
        rs += __shfl_xor(rs, 4, 32);
        rs += __shfl_xor(rs, 8, 32);
        lrow[r] = lrow[r] * alpha[r] + rs;
        mrow[r] = mnew[r];
        int prow = r + 8 * half;
        Pw[wave][prow][l16]      = (__bf16)p0;
        Pw[wave][prow][16 + l16] = (__bf16)p1;
#pragma unroll
        for (int d = 0; d < 4; ++d) o[d][r] *= alpha[r];
      }
      // Same-wave LDS store->load: DS is in-order per wave; drain our stores.
      asm volatile("s_wait_dscnt 0x0" ::: "memory");

      const uint32_t* prow_u = (const uint32_t*)&Pw[wave][l16][0];
      v16bf pf = load_afrag(prow_u, half);
#pragma unroll
      for (int d = 0; d < 4; ++d) {
        v16bf vf =
            load_bfrag((const uint32_t*)&VsT[buf][d * 16 + l16][0], half);
        o[d] = wmma_bf16(pf, vf, o[d]);
      }
    }
  }

  // Epilogue: O /= l, store bf16 as [B,L,H,Dh] (rows = M, cols = h*64+e)
  const int b = bh >> 4, h = bh & (H_ - 1);
#pragma unroll
  for (int r = 0; r < 8; ++r) {
    float inv = 1.0f / lrow[r];
    int rowg = q0 + r + 8 * half;
    size_t rb = (((size_t)b * L_ + rowg) * H_ + h) * DH_;
#pragma unroll
    for (int d = 0; d < 4; ++d)
      ob[rb + d * 16 + l16] = (__bf16)(o[d][r] * inv);
  }
}

// ---------------------------------------------------------------------------
// Kernel 3: output projection. out[4096,1024] f32 = O[4096,1024]bf16 * Wout.
// A tile is a raw bf16 copy -> staged with async global->LDS.
// ---------------------------------------------------------------------------
__global__ __launch_bounds__(256) void outproj_kernel(
    const __bf16* __restrict__ ob, const float* __restrict__ Wo,
    float* __restrict__ out) {
  const int N0 = blockIdx.x * 64;
  const int M0 = blockIdx.y * 128;
  const int tid  = threadIdx.x;
  const int lane = tid & 31, wave = tid >> 5;
  const int half = lane >> 4, l16 = lane & 15;
  const int m0w = (wave >> 1) * 32;
  const int n0w = (wave & 1) * 32;

  __shared__ uint32_t As[128][16];
  __shared__ __bf16   WsT[64][32];

  v8f acc[2][2];
  acc[0][0] = zero8(); acc[0][1] = zero8();
  acc[1][0] = zero8(); acc[1][1] = zero8();

  for (int kt = 0; kt < D_; kt += 32) {
    // A tile: bf16, already pair-packed in memory -> async b128 copies
#pragma unroll
    for (int i = 0; i < 2; ++i) {
      int idx  = tid + i * 256;
      int rowm = idx >> 2;
      int col8 = (idx & 3) * 8;
      async_copy_b128(&As[rowm][col8 >> 1],
                      ob + (size_t)(M0 + rowm) * D_ + kt + col8);
    }
    // W tile transposed (f32 -> bf16)
#pragma unroll
    for (int i = 0; i < 2; ++i) {
      int idx4 = tid + i * 256;
      int kk   = idx4 >> 4;
      int n    = (idx4 & 15) * 4;
      float4 f = *(const float4*)(Wo + (size_t)(kt + kk) * D_ + N0 + n);
      WsT[n + 0][kk] = (__bf16)f.x;
      WsT[n + 1][kk] = (__bf16)f.y;
      WsT[n + 2][kk] = (__bf16)f.z;
      WsT[n + 3][kk] = (__bf16)f.w;
    }
    async_wait0();
    __syncthreads();

    v16bf a0 = load_afrag(&As[m0w + l16][0], half);
    v16bf a1 = load_afrag(&As[m0w + 16 + l16][0], half);
    v16bf b0 = load_bfrag((const uint32_t*)&WsT[n0w + l16][0], half);
    v16bf b1 = load_bfrag((const uint32_t*)&WsT[n0w + 16 + l16][0], half);
    acc[0][0] = wmma_bf16(a0, b0, acc[0][0]);
    acc[0][1] = wmma_bf16(a0, b1, acc[0][1]);
    acc[1][0] = wmma_bf16(a1, b0, acc[1][0]);
    acc[1][1] = wmma_bf16(a1, b1, acc[1][1]);
    __syncthreads();
  }

#pragma unroll
  for (int i = 0; i < 2; ++i)
#pragma unroll
    for (int jt = 0; jt < 2; ++jt)
#pragma unroll
      for (int r = 0; r < 8; ++r) {
        int gm = M0 + m0w + i * 16 + r + 8 * half;
        int gn = N0 + n0w + jt * 16 + l16;
        out[(size_t)gm * D_ + gn] = acc[i][jt][r];
      }
}

// ---------------------------------------------------------------------------
// Host launcher
// ---------------------------------------------------------------------------
extern "C" void kernel_launch(void* const* d_in, const int* in_sizes, int n_in,
                              void* d_out, int out_size, void* d_ws,
                              size_t ws_size, hipStream_t stream) {
  (void)in_sizes; (void)n_in; (void)out_size; (void)ws_size;
  const float* x  = (const float*)d_in[0];
  const float* Wq = (const float*)d_in[1];
  const float* Wk = (const float*)d_in[2];
  const float* Wv = (const float*)d_in[3];
  const float* Wo = (const float*)d_in[4];
  float* out = (float*)d_out;

  const size_t nqkv = (size_t)B_ * H_ * L_ * DH_;   // 4,194,304 elems (8 MB bf16)
  char* ws = (char*)d_ws;
  __bf16* qb = (__bf16*)(ws + 0 * nqkv * sizeof(__bf16));
  __bf16* kb = (__bf16*)(ws + 1 * nqkv * sizeof(__bf16));
  __bf16* vb = (__bf16*)(ws + 2 * nqkv * sizeof(__bf16));
  __bf16* ob = (__bf16*)(ws + 3 * nqkv * sizeof(__bf16));

  dim3 blk(256);
  qkv_kernel<<<dim3(D_ / 64, M_ / 128, 3), blk, 0, stream>>>(x, Wq, Wk, Wv,
                                                             qb, kb, vb);
  attn_kernel<<<dim3(L_ / 128, B_ * H_), blk, 0, stream>>>(qb, kb, vb, ob);
  outproj_kernel<<<dim3(D_ / 64, M_ / 128), blk, 0, stream>>>(ob, Wo, out);
}
